// ElevationTransformer_NO6_contiEm_AT_63256278336222
// MI455X (gfx1250) — matrix operand order repaired and checked by reading, not compiled, not run on hardware
//
#include <hip/hip_runtime.h>
#include <hip/hip_bf16.h>
#include <cstddef>

// ---------------------------------------------------------------------------
// Problem constants (ViT-Tiny + SAT fusion, B=128)
// ---------------------------------------------------------------------------
#define BATCH   128
#define SEQ     197
#define DIM     192
#define NHEAD   3
#define HD      64              // DIM / NHEAD
#define NPATCH  196
#define KPATCH  768             // 3*16*16
#define MLPD    768
#define QKVN    576
#define SATN    256             // 8 heads * 32
#define DEPTH   12

#define M_TOK   (BATCH * SEQ)       // 25216  (= 197*128)
#define M_SAT   (M_TOK * 2)         // 50432  (= 394*128)
#define M_PATCH (BATCH * NPATCH)    // 25088  (= 196*128)

typedef _Float16 v16h __attribute__((ext_vector_type(16)));
typedef _Float16 v8h  __attribute__((ext_vector_type(8)));
typedef float    v8f  __attribute__((ext_vector_type(8)));

#define WMMA_F16(A, B, C) \
    __builtin_amdgcn_wmma_f32_16x16x32_f16(false, (A), false, (B), (short)0, (C), false, false)

#define CAT16(lo, hi) \
    __builtin_shufflevector((lo), (hi), 0, 1, 2, 3, 4, 5, 6, 7, 8, 9, 10, 11, 12, 13, 14, 15)

__device__ __forceinline__ float gelu_exact(float x) {
    return 0.5f * x * (1.0f + erff(x * 0.70710678118654752f));
}

// ---------------------------------------------------------------------------
// WMMA GEMM: C[M,N] = epilogue(A[M,K] @ W[K,N] + bias)
//   Block tile 128x64, K-chunk 32, 8 wave32 in a 4x2 grid, 32x32 register
//   tile per wave (4 v_wmma per chunk, 2xA/2xB reuse), double-buffered LDS
//   in fragment-ready layouts (A permuted-contiguous, B column-major).
//   M%128==0, N%64==0, K%32==0 for every call => no bounds checks.
// ---------------------------------------------------------------------------
#define ASUB_STRIDE 768          // 32 lanes * 24 halves
#define BCOL_STRIDE 40

template <typename OutT, int EPI>
__global__ __launch_bounds__(256) void gemm_wmma_f16(
    const _Float16* __restrict__ A, const _Float16* __restrict__ W,
    const float* __restrict__ bias, const float* __restrict__ resid,
    OutT* __restrict__ C, int M, int N, int K)
{
    __shared__ alignas(16) _Float16 Ash[2][8 * ASUB_STRIDE];   // 24KB
    __shared__ alignas(16) _Float16 Bsh[2][64 * BCOL_STRIDE];  // 10KB

    const int tid  = threadIdx.x;
    const int lane = tid & 31;
    const int wave = tid >> 5;
    const int wm   = wave >> 1;       // 0..3  -> M offset wm*32
    const int wn   = wave & 1;        // 0..1  -> N offset wn*32
    const int half = lane >> 4;
    const int l15  = lane & 15;
    const int bM   = blockIdx.y * 128;
    const int bN   = blockIdx.x * 64;

    // A staging: thread loads row ar, 16 contiguous halves at col acb
    const int ar   = tid >> 1;
    const int acb  = (tid & 1) * 16;
    const int asub = ar >> 4;
    const int am   = ar & 15;
    const int aoff0 = asub * ASUB_STRIDE + (am + (acb >> 4) * 16) * 24;
    const int aoff1 = aoff0 + 8;

    // B staging: thread loads K-row bk, 8 halves at col bcb
    const int bk  = tid >> 3;
    const int bcb = (tid & 7) * 8;

    // fragment read offsets (halves)
    const int fa0 = (wm * 2 + 0) * ASUB_STRIDE + lane * 24;
    const int fa1 = (wm * 2 + 1) * ASUB_STRIDE + lane * 24;
    const int fb0 = (wn * 32 + l15) * BCOL_STRIDE + half * 16;
    const int fb1 = (wn * 32 + 16 + l15) * BCOL_STRIDE + half * 16;

    {   // prologue: stage chunk 0 into buffer 0
        const _Float16* asrc = A + (size_t)(bM + ar) * K + acb;
        *(v8h*)&Ash[0][aoff0] = *(const v8h*)asrc;
        *(v8h*)&Ash[0][aoff1] = *(const v8h*)(asrc + 8);
        const v8h bvec = *(const v8h*)(W + (size_t)bk * N + bN + bcb);
        #pragma unroll
        for (int j = 0; j < 8; ++j)
            Bsh[0][(bcb + j) * BCOL_STRIDE + bk] = bvec[j];
    }
    __syncthreads();

    v8f acc00 = {}, acc01 = {}, acc10 = {}, acc11 = {};
    int cur = 0;
    for (int k0 = 0; k0 < K; k0 += 32) {
        const bool more = (k0 + 32 < K);
        v8h na0, na1, nb;
        if (more) {
            const _Float16* asrc = A + (size_t)(bM + ar) * K + (k0 + 32) + acb;
            na0 = *(const v8h*)asrc;
            na1 = *(const v8h*)(asrc + 8);
            nb  = *(const v8h*)(W + (size_t)(k0 + 32 + bk) * N + bN + bcb);
            if (k0 + 64 < K) {
                __builtin_prefetch(A + (size_t)(bM + ar) * K + (k0 + 64) + acb, 0, 3);
                __builtin_prefetch(W + (size_t)(k0 + 64 + bk) * N + bN + bcb, 0, 3);
            }
        }

        const _Float16* As = Ash[cur];
        const _Float16* Bs = Bsh[cur];
        const v16h a0 = *(const v16h*)(As + fa0);
        const v16h a1 = *(const v16h*)(As + fa1);
        const v16h b0 = *(const v16h*)(Bs + fb0);
        const v16h b1 = *(const v16h*)(Bs + fb1);

        acc00 = WMMA_F16(a0, b0, acc00);
        acc01 = WMMA_F16(a0, b1, acc01);
        acc10 = WMMA_F16(a1, b0, acc10);
        acc11 = WMMA_F16(a1, b1, acc11);

        if (more) {
            _Float16* Ad = Ash[cur ^ 1];
            _Float16* Bd = Bsh[cur ^ 1];
            *(v8h*)&Ad[aoff0] = na0;
            *(v8h*)&Ad[aoff1] = na1;
            #pragma unroll
            for (int j = 0; j < 8; ++j)
                Bd[(bcb + j) * BCOL_STRIDE + bk] = nb[j];
        }
        __syncthreads();
        cur ^= 1;
    }

    // epilogue (C/D layout: lane -> column, element i -> row, +8 hi lanes)
    const int col0 = bN + wn * 32 + l15;
    const int col1 = col0 + 16;
    const float bv0 = bias[col0];
    const float bv1 = bias[col1];
    #pragma unroll
    for (int i = 0; i < 8; ++i) {
        const int r0 = bM + wm * 32 + half * 8 + i;
        const int r1 = r0 + 16;
        float e00 = acc00[i] + bv0, e01 = acc01[i] + bv1;
        float e10 = acc10[i] + bv0, e11 = acc11[i] + bv1;
        if constexpr (EPI == 1) {
            e00 += resid[(size_t)r0 * N + col0];
            e01 += resid[(size_t)r0 * N + col1];
            e10 += resid[(size_t)r1 * N + col0];
            e11 += resid[(size_t)r1 * N + col1];
        }
        if constexpr (EPI == 2) {
            e00 = gelu_exact(e00); e01 = gelu_exact(e01);
            e10 = gelu_exact(e10); e11 = gelu_exact(e11);
        }
        C[(size_t)r0 * N + col0] = (OutT)e00;
        C[(size_t)r0 * N + col1] = (OutT)e01;
        C[(size_t)r1 * N + col0] = (OutT)e10;
        C[(size_t)r1 * N + col1] = (OutT)e11;
    }
}

// ---------------------------------------------------------------------------
// WMMA block attention, one (batch, head) per workgroup.
//   qkv f16 [M_TOK,576] rows b*SEQ+s, layout [q|k|v]; out f16 [M_TOK,192].
//   Keys padded to 224 (14 tiles / 7 chunks). LDS:
//     Kst[key][d]  = B-fragment layout for Q*K^T  (28.0 KB)
//     Vt [d][key]  = B-fragment layout for P*V    (28.0 KB)
//     Sc [16][232] = f16 score rows; row-major IS the A-frag layout for P*V
//   13 query groups of 16; per group: 28 wmma (scores) + 28 wmma (P*V).
// ---------------------------------------------------------------------------
#define SPAD 224
#define SC_STRIDE 232

__global__ __launch_bounds__(256) void attn_wmma_kernel(
    const _Float16* __restrict__ qkv, _Float16* __restrict__ out)
{
    __shared__ alignas(16) _Float16 Kst[SPAD * 64];
    __shared__ alignas(16) _Float16 Vt[64 * SPAD];
    __shared__ alignas(16) _Float16 Sc[16 * SC_STRIDE];

    const int b    = blockIdx.x / NHEAD;
    const int h    = blockIdx.x % NHEAD;
    const int tid  = threadIdx.x;
    const int lane = tid & 31;
    const int wave = tid >> 5;
    const int half = lane >> 4;
    const int l15  = lane & 15;
    const int kb   = half * 8;
    const size_t rowbase = (size_t)b * SEQ;

    // stage K ([key][d]) and V transposed ([d][key]); zero-pad keys >= SEQ
    for (int i = tid; i < (SPAD * 64) / 8; i += 256) {
        const int key  = i >> 3;
        const int dseg = (i & 7) * 8;
        v8h kv = {}, vv = {};
        if (key < SEQ) {
            const _Float16* src = qkv + (rowbase + key) * QKVN + h * HD + dseg;
            kv = *(const v8h*)(src + DIM);
            vv = *(const v8h*)(src + 2 * DIM);
        }
        *(v8h*)&Kst[key * 64 + dseg] = kv;
        #pragma unroll
        for (int j = 0; j < 8; ++j) Vt[(dseg + j) * SPAD + key] = vv[j];
    }
    __syncthreads();

    for (int qg = 0; qg < 13; ++qg) {
        // Q A-fragments straight from global (rows clamped; extras unused)
        const int qrow = qg * 16 + l15;
        const int qcl  = (qrow < SEQ) ? qrow : (SEQ - 1);
        const _Float16* qptr = qkv + (rowbase + qcl) * QKVN + h * HD;
        const v16h qa0 = CAT16(*(const v8h*)(qptr + half * 16),
                               *(const v8h*)(qptr + half * 16 + 8));
        const v16h qa1 = CAT16(*(const v8h*)(qptr + 32 + half * 16),
                               *(const v8h*)(qptr + 32 + half * 16 + 8));

        // scores: wave w covers key-tiles w and w+8
        #pragma unroll
        for (int tt = 0; tt < 2; ++tt) {
            const int t = wave + tt * 8;
            if (t < 14) {
                const v16h kb0 = *(const v16h*)&Kst[(t * 16 + l15) * 64 + half * 16];
                const v16h kb1 = *(const v16h*)&Kst[(t * 16 + l15) * 64 + 32 + half * 16];
                v8f acc = {};
                acc = WMMA_F16(qa0, kb0, acc);
                acc = WMMA_F16(qa1, kb1, acc);
                #pragma unroll
                for (int i = 0; i < 8; ++i)
                    Sc[(half * 8 + i) * SC_STRIDE + t * 16 + l15] =
                        (_Float16)(acc[i] * 0.125f);       // hd^-0.5
            }
        }
        __syncthreads();

        // softmax in place (mask keys >= SEQ); wave handles rows 2w, 2w+1
        #pragma unroll
        for (int rr = 0; rr < 2; ++rr) {
            const int r = wave * 2 + rr;
            float sv[7];
            float mx = -1e30f;
            #pragma unroll
            for (int j = 0; j < 7; ++j) {
                const int key = lane + 32 * j;
                sv[j] = (key < SEQ) ? (float)Sc[r * SC_STRIDE + key] : -1e30f;
                mx = fmaxf(mx, sv[j]);
            }
            #pragma unroll
            for (int m = 16; m >= 1; m >>= 1) mx = fmaxf(mx, __shfl_xor(mx, m, 32));
            float pv[7], sum = 0.0f;
            #pragma unroll
            for (int j = 0; j < 7; ++j) {
                const int key = lane + 32 * j;
                pv[j] = (key < SEQ) ? __expf(sv[j] - mx) : 0.0f;
                sum += pv[j];
            }
            #pragma unroll
            for (int m = 16; m >= 1; m >>= 1) sum += __shfl_xor(sum, m, 32);
            const float inv = 1.0f / sum;
            #pragma unroll
            for (int j = 0; j < 7; ++j)
                Sc[r * SC_STRIDE + lane + 32 * j] = (_Float16)(pv[j] * inv);
        }
        __syncthreads();

        // P*V: waves 0..3 each produce one 16-wide d-subtile of 16x64
        if (wave < 4) {
            v8f acc = {};
            #pragma unroll
            for (int c = 0; c < 7; ++c) {
                const v16h pa = CAT16(*(const v8h*)&Sc[l15 * SC_STRIDE + c * 32 + half * 16],
                                      *(const v8h*)&Sc[l15 * SC_STRIDE + c * 32 + half * 16 + 8]);
                const v16h vb = *(const v16h*)&Vt[(wave * 16 + l15) * SPAD
                                                  + c * 32 + half * 16];
                acc = WMMA_F16(pa, vb, acc);
            }
            #pragma unroll
            for (int i = 0; i < 8; ++i) {
                const int row = qg * 16 + half * 8 + i;
                if (row < SEQ)
                    out[(rowbase + row) * DIM + h * HD + wave * 16 + l15] =
                        (_Float16)acc[i];
            }
        }
        __syncthreads();
    }
}

// ---------------------------------------------------------------------------
// fp32 -> f16 weight conversion (and transposed variant for patch_w)
// ---------------------------------------------------------------------------
__global__ void f32_to_f16_kernel(const float* __restrict__ a,
                                  _Float16* __restrict__ o, int n) {
    for (int i = blockIdx.x * blockDim.x + threadIdx.x; i < n;
         i += gridDim.x * blockDim.x)
        o[i] = (_Float16)a[i];
}

__global__ void transpose_cvt_kernel(const float* __restrict__ a,  // [R,C]
                                     _Float16* __restrict__ o,     // [C,R]
                                     int R, int C) {
    int i = blockIdx.x * blockDim.x + threadIdx.x;
    if (i >= R * C) return;
    int r = i / C, c = i % C;
    o[(size_t)c * R + r] = (_Float16)a[i];
}

// ---------------------------------------------------------------------------
// im2col for the 16x16/stride-16 patch conv: Ap[b*196+p][c*256+ky*16+kx]
// ---------------------------------------------------------------------------
__global__ __launch_bounds__(256) void im2col_kernel(const float* __restrict__ x,
                                                     _Float16* __restrict__ ap) {
    int idx = blockIdx.x * blockDim.x + threadIdx.x;    // exact: MP*KPATCH
    int k = idx % KPATCH;
    int p = (idx / KPATCH) % NPATCH;
    int b = idx / (KPATCH * NPATCH);
    int c = k >> 8, rem = k & 255, ky = rem >> 4, kx = rem & 15;
    int py = p / 14, px = p % 14;
    ap[idx] = (_Float16)x[(((size_t)b * 3 + c) * 224 + (py * 16 + ky)) * 224
                          + (px * 16 + kx)];
}

// ---------------------------------------------------------------------------
// Elevation weight: mean over 56x56, then bisect over linspace(0.1,1.0,10)
// ---------------------------------------------------------------------------
__global__ __launch_bounds__(256) void eleva_weight_kernel(
    const float* __restrict__ xg, float* __restrict__ wout) {
    __shared__ float red[8];
    const int b = blockIdx.x, tid = threadIdx.x;
    float s = 0.0f;
    for (int i = tid; i < 3136; i += 256) s += xg[(size_t)b * 3136 + i];
    #pragma unroll
    for (int m = 16; m >= 1; m >>= 1) s += __shfl_xor(s, m, 32);
    if ((tid & 31) == 0) red[tid >> 5] = s;
    __syncthreads();
    if (tid == 0) {
        float tot = 0.0f;
        #pragma unroll
        for (int i = 0; i < 8; ++i) tot += red[i];
        const float mean = tot / 3136.0f;
        int idx = 0;
        while (idx < 10 && 0.1f * (float)(idx + 1) < mean) ++idx;   // searchsorted left
        if (idx > 9) idx = 9;
        const float per_hi = 0.1f * (float)(idx + 1);
        const float per_lo = (idx == 0) ? 0.0f : 0.1f * (float)idx;
        const float idx_f = (float)idx;
        const float idx_inter = idx_f - (per_hi - mean) / (per_hi - per_lo);
        wout[b] = ((idx == 9) ? idx_f : idx_inter) * 0.1f;
    }
}

// ---------------------------------------------------------------------------
// Token assembly: tok = [cls | patches] + pos ; at_in slot0 = tok,
// slot1 = eleva_embed * w[b]   ->  f16 [M_SAT, DIM]
// ---------------------------------------------------------------------------
__global__ __launch_bounds__(256) void build_at_in_kernel(
    const float* __restrict__ tokC, const float* __restrict__ cls,
    const float* __restrict__ pos, const float* __restrict__ eleva,
    const float* __restrict__ wv, _Float16* __restrict__ at) {
    int idx = blockIdx.x * blockDim.x + threadIdx.x;    // exact: BATCH*SEQ*DIM
    int d = idx % DIM;
    int s = (idx / DIM) % SEQ;
    int b = idx / (DIM * SEQ);
    float tok = ((s == 0) ? cls[d]
                          : tokC[((size_t)b * NPATCH + (s - 1)) * DIM + d])
                + pos[(size_t)s * DIM + d];
    size_t r2 = ((size_t)b * SEQ + s) * 2;
    at[r2 * DIM + d]       = (_Float16)tok;
    at[(r2 + 1) * DIM + d] = (_Float16)(eleva[d] * wv[b]);
}

// ---------------------------------------------------------------------------
// SAT: per (b,s,head) 2x2 attention over (token, elevation) slots
// ---------------------------------------------------------------------------
__global__ __launch_bounds__(256) void sat_combine_kernel(
    const _Float16* __restrict__ q, const _Float16* __restrict__ k,
    const _Float16* __restrict__ v, _Float16* __restrict__ ctx) {
    int t = blockIdx.x * blockDim.x + threadIdx.x;      // exact: M_TOK*8
    int h = t & 7;
    int bs = t >> 3;
    size_t b0 = ((size_t)bs * 2) * SATN + h * 32;
    size_t b1 = ((size_t)bs * 2 + 1) * SATN + h * 32;

    float q0[32], q1[32], k0[32], k1[32], v0[32], v1[32];
    #pragma unroll
    for (int d = 0; d < 32; ++d) {
        q0[d] = (float)q[b0 + d]; q1[d] = (float)q[b1 + d];
        k0[d] = (float)k[b0 + d]; k1[d] = (float)k[b1 + d];
        v0[d] = (float)v[b0 + d]; v1[d] = (float)v[b1 + d];
    }
    float s00 = 0, s01 = 0, s10 = 0, s11 = 0;
    #pragma unroll
    for (int d = 0; d < 32; ++d) {
        s00 += q0[d] * k0[d]; s01 += q0[d] * k1[d];
        s10 += q1[d] * k0[d]; s11 += q1[d] * k1[d];
    }
    const float sc = 0.17677669529663687f;  // 1/sqrt(32)
    s00 *= sc; s01 *= sc; s10 *= sc; s11 *= sc;
    float m0 = fmaxf(s00, s01), m1 = fmaxf(s10, s11);
    float e00 = __expf(s00 - m0), e01 = __expf(s01 - m0);
    float e10 = __expf(s10 - m1), e11 = __expf(s11 - m1);
    float a00 = e00 / (e00 + e01), a01 = e01 / (e00 + e01);
    float a10 = e10 / (e10 + e11), a11 = e11 / (e10 + e11);
    #pragma unroll
    for (int d = 0; d < 32; ++d) {
        ctx[b0 + d] = (_Float16)(a00 * v0[d] + a01 * v1[d]);
        ctx[b1 + d] = (_Float16)(a10 * v0[d] + a11 * v1[d]);
    }
}

// h = sat_o[:, :, 0, :] + sat_o[:, :, 1, :]  -> fp32 residual stream
__global__ __launch_bounds__(256) void sat_sum_kernel(
    const _Float16* __restrict__ sato, float* __restrict__ z) {
    int idx = blockIdx.x * blockDim.x + threadIdx.x;    // exact: M_TOK*DIM
    int d = idx % DIM;
    int r = idx / DIM;
    z[idx] = (float)sato[((size_t)2 * r) * DIM + d]
           + (float)sato[((size_t)2 * r + 1) * DIM + d];
}

// ---------------------------------------------------------------------------
// LayerNorm over D=192 — one wave32 per token row, 6 elems/lane, shfl reduce.
// ---------------------------------------------------------------------------
__global__ __launch_bounds__(256) void layernorm_kernel(
    const float* __restrict__ z, const float* __restrict__ g,
    const float* __restrict__ b, _Float16* __restrict__ y, int M) {
    const int row = blockIdx.x * 8 + (threadIdx.x >> 5);
    const int lane = threadIdx.x & 31;
    if (row >= M) return;
    const float* zr = z + (size_t)row * DIM;
    float vals[6], s = 0.0f;
    #pragma unroll
    for (int i = 0; i < 6; ++i) { vals[i] = zr[lane + 32 * i]; s += vals[i]; }
    #pragma unroll
    for (int m = 16; m >= 1; m >>= 1) s += __shfl_xor(s, m, 32);
    const float mean = s * (1.0f / DIM);
    float vs = 0.0f;
    #pragma unroll
    for (int i = 0; i < 6; ++i) { float d = vals[i] - mean; vs += d * d; }
    #pragma unroll
    for (int m = 16; m >= 1; m >>= 1) vs += __shfl_xor(vs, m, 32);
    const float rstd = rsqrtf(vs * (1.0f / DIM) + 1e-5f);
    #pragma unroll
    for (int i = 0; i < 6; ++i) {
        int d = lane + 32 * i;
        y[(size_t)row * DIM + d] =
            (_Float16)((vals[i] - mean) * rstd * g[d] + b[d]);
    }
}

// Final LN on the cls token only -> fp32 d_out[b, DIM]
__global__ __launch_bounds__(32) void final_ln_cls_kernel(
    const float* __restrict__ z, const float* __restrict__ g,
    const float* __restrict__ b, float* __restrict__ out) {
    const int bb = blockIdx.x, lane = threadIdx.x;
    const float* zr = z + (size_t)bb * SEQ * DIM;       // cls row = b*SEQ
    float vals[6], s = 0.0f;
    #pragma unroll
    for (int i = 0; i < 6; ++i) { vals[i] = zr[lane + 32 * i]; s += vals[i]; }
    #pragma unroll
    for (int m = 16; m >= 1; m >>= 1) s += __shfl_xor(s, m, 32);
    const float mean = s * (1.0f / DIM);
    float vs = 0.0f;
    #pragma unroll
    for (int i = 0; i < 6; ++i) { float d = vals[i] - mean; vs += d * d; }
    #pragma unroll
    for (int m = 16; m >= 1; m >>= 1) vs += __shfl_xor(vs, m, 32);
    const float rstd = rsqrtf(vs * (1.0f / DIM) + 1e-5f);
    #pragma unroll
    for (int i = 0; i < 6; ++i) {
        int d = lane + 32 * i;
        out[(size_t)bb * DIM + d] = (vals[i] - mean) * rstd * g[d] + b[d];
    }
}

// ---------------------------------------------------------------------------
// Host orchestration
// ---------------------------------------------------------------------------
extern "C" void kernel_launch(void* const* d_in, const int* in_sizes, int n_in,
                              void* d_out, int out_size, void* d_ws, size_t ws_size,
                              hipStream_t stream) {
    const float* x        = (const float*)d_in[0];
    const float* x_g_norm = (const float*)d_in[1];
    const float* patch_w  = (const float*)d_in[2];
    const float* patch_b  = (const float*)d_in[3];
    const float* cls_tok  = (const float*)d_in[4];
    const float* pos_emb  = (const float*)d_in[5];
    const float* eleva    = (const float*)d_in[6];
    const float* sat_wq   = (const float*)d_in[7];
    const float* sat_bq   = (const float*)d_in[8];
    const float* sat_wk   = (const float*)d_in[9];
    const float* sat_bk   = (const float*)d_in[10];
    const float* sat_wv   = (const float*)d_in[11];
    const float* sat_bv   = (const float*)d_in[12];
    const float* sat_wo   = (const float*)d_in[13];
    const float* sat_bo   = (const float*)d_in[14];
    const float* ln1_g    = (const float*)d_in[15];
    const float* ln1_b    = (const float*)d_in[16];
    const float* qkv_w    = (const float*)d_in[17];
    const float* qkv_b    = (const float*)d_in[18];
    const float* proj_w   = (const float*)d_in[19];
    const float* proj_b   = (const float*)d_in[20];
    const float* ln2_g    = (const float*)d_in[21];
    const float* ln2_b    = (const float*)d_in[22];
    const float* fc1_w    = (const float*)d_in[23];
    const float* fc1_b    = (const float*)d_in[24];
    const float* fc2_w    = (const float*)d_in[25];
    const float* fc2_b    = (const float*)d_in[26];
    const float* norm_g   = (const float*)d_in[27];
    const float* norm_b   = (const float*)d_in[28];

    // bump allocator over d_ws (256B aligned)
    char* wsb = (char*)d_ws;
    size_t off = 0;
    auto alloc = [&](size_t bytes) -> void* {
        off = (off + 255) & ~(size_t)255;
        void* p = wsb + off;
        off += bytes;
        return p;
    };

    _Float16* w_patch = (_Float16*)alloc((size_t)KPATCH * DIM * 2);
    _Float16* w_satq  = (_Float16*)alloc((size_t)DIM * SATN * 2);
    _Float16* w_satk  = (_Float16*)alloc((size_t)DIM * SATN * 2);
    _Float16* w_satv  = (_Float16*)alloc((size_t)DIM * SATN * 2);
    _Float16* w_sato  = (_Float16*)alloc((size_t)SATN * DIM * 2);
    _Float16* w_qkv   = (_Float16*)alloc((size_t)DEPTH * DIM * QKVN * 2);
    _Float16* w_proj  = (_Float16*)alloc((size_t)DEPTH * DIM * DIM * 2);
    _Float16* w_fc1   = (_Float16*)alloc((size_t)DEPTH * DIM * MLPD * 2);
    _Float16* w_fc2   = (_Float16*)alloc((size_t)DEPTH * MLPD * DIM * 2);
    float*    wvec    = (float*)alloc(BATCH * 4);
    _Float16* Ap      = (_Float16*)alloc((size_t)M_PATCH * KPATCH * 2);
    float*    tokC    = (float*)alloc((size_t)M_PATCH * DIM * 4);
    _Float16* at_in   = (_Float16*)alloc((size_t)M_SAT * DIM * 2);
    _Float16* qs      = (_Float16*)alloc((size_t)M_SAT * SATN * 2);
    _Float16* ks      = (_Float16*)alloc((size_t)M_SAT * SATN * 2);
    _Float16* vs      = (_Float16*)alloc((size_t)M_SAT * SATN * 2);
    _Float16* ctx     = (_Float16*)alloc((size_t)M_SAT * SATN * 2);
    _Float16* sato    = (_Float16*)alloc((size_t)M_SAT * DIM * 2);
    float*    z       = (float*)alloc((size_t)M_TOK * DIM * 4);
    _Float16* y       = (_Float16*)alloc((size_t)M_TOK * DIM * 2);
    _Float16* qkvb    = (_Float16*)alloc((size_t)M_TOK * QKVN * 2);
    _Float16* attnb   = (_Float16*)alloc((size_t)M_TOK * DIM * 2);
    _Float16* mlpb    = (_Float16*)alloc((size_t)M_TOK * MLPD * 2);
    (void)ws_size; (void)n_in; (void)in_sizes; (void)out_size;

    // ---- weight conversion to f16 (per-launch; deterministic) ----
    auto cvt = [&](const float* src, _Float16* dst, int n) {
        f32_to_f16_kernel<<<(n + 255) / 256, 256, 0, stream>>>(src, dst, n);
    };
    transpose_cvt_kernel<<<(DIM * KPATCH + 255) / 256, 256, 0, stream>>>(
        patch_w, w_patch, DIM, KPATCH);                 // (192,768) -> [768,192]
    cvt(sat_wq, w_satq, DIM * SATN);
    cvt(sat_wk, w_satk, DIM * SATN);
    cvt(sat_wv, w_satv, DIM * SATN);
    cvt(sat_wo, w_sato, SATN * DIM);
    cvt(qkv_w, w_qkv, DEPTH * DIM * QKVN);
    cvt(proj_w, w_proj, DEPTH * DIM * DIM);
    cvt(fc1_w, w_fc1, DEPTH * DIM * MLPD);
    cvt(fc2_w, w_fc2, DEPTH * MLPD * DIM);

    // ---- elevation weight per sample ----
    eleva_weight_kernel<<<BATCH, 256, 0, stream>>>(x_g_norm, wvec);

    // ---- patch embed: im2col + WMMA GEMM ----
    im2col_kernel<<<(M_PATCH * KPATCH) / 256, 256, 0, stream>>>(x, Ap);
    gemm_wmma_f16<float, 0><<<dim3(DIM / 64, M_PATCH / 128), 256, 0, stream>>>(
        Ap, w_patch, patch_b, nullptr, tokC, M_PATCH, DIM, KPATCH);

    // ---- tokens + pos + SAT input assembly ----
    build_at_in_kernel<<<(M_TOK * DIM) / 256, 256, 0, stream>>>(
        tokC, cls_tok, pos_emb, eleva, wvec, at_in);

    // ---- SAT q/k/v projections (WMMA), 2x2 attention, out projection ----
    gemm_wmma_f16<_Float16, 0><<<dim3(SATN / 64, M_SAT / 128), 256, 0, stream>>>(
        at_in, w_satq, sat_bq, nullptr, qs, M_SAT, SATN, DIM);
    gemm_wmma_f16<_Float16, 0><<<dim3(SATN / 64, M_SAT / 128), 256, 0, stream>>>(
        at_in, w_satk, sat_bk, nullptr, ks, M_SAT, SATN, DIM);
    gemm_wmma_f16<_Float16, 0><<<dim3(SATN / 64, M_SAT / 128), 256, 0, stream>>>(
        at_in, w_satv, sat_bv, nullptr, vs, M_SAT, SATN, DIM);
    sat_combine_kernel<<<(M_TOK * 8) / 256, 256, 0, stream>>>(qs, ks, vs, ctx);
    gemm_wmma_f16<_Float16, 0><<<dim3(DIM / 64, M_SAT / 128), 256, 0, stream>>>(
        ctx, w_sato, sat_bo, nullptr, sato, M_SAT, DIM, SATN);
    sat_sum_kernel<<<(M_TOK * DIM) / 256, 256, 0, stream>>>(sato, z);

    // ---- 12 transformer blocks ----
    for (int l = 0; l < DEPTH; ++l) {
        layernorm_kernel<<<M_TOK / 8, 256, 0, stream>>>(
            z, ln1_g + l * DIM, ln1_b + l * DIM, y, M_TOK);
        gemm_wmma_f16<_Float16, 0><<<dim3(QKVN / 64, M_TOK / 128), 256, 0, stream>>>(
            y, w_qkv + (size_t)l * DIM * QKVN, qkv_b + l * QKVN, nullptr,
            qkvb, M_TOK, QKVN, DIM);
        attn_wmma_kernel<<<BATCH * NHEAD, 256, 0, stream>>>(qkvb, attnb);
        gemm_wmma_f16<float, 1><<<dim3(DIM / 64, M_TOK / 128), 256, 0, stream>>>(
            attnb, w_proj + (size_t)l * DIM * DIM, proj_b + l * DIM, z,
            z, M_TOK, DIM, DIM);
        layernorm_kernel<<<M_TOK / 8, 256, 0, stream>>>(
            z, ln2_g + l * DIM, ln2_b + l * DIM, y, M_TOK);
        gemm_wmma_f16<_Float16, 2><<<dim3(MLPD / 64, M_TOK / 128), 256, 0, stream>>>(
            y, w_fc1 + (size_t)l * DIM * MLPD, fc1_b + l * MLPD, nullptr,
            mlpb, M_TOK, MLPD, DIM);
        gemm_wmma_f16<float, 1><<<dim3(DIM / 64, M_TOK / 128), 256, 0, stream>>>(
            mlpb, w_fc2 + (size_t)l * MLPD * DIM, fc2_b + l * DIM, z,
            z, M_TOK, DIM, MLPD);
    }

    // ---- final LN on cls tokens -> d_out [B, D] fp32 ----
    final_ln_cls_kernel<<<BATCH, 32, 0, stream>>>(z, norm_g, norm_b,
                                                  (float*)d_out);
}
